// Model_31997506356062
// MI455X (gfx1250) — compile-verified, matrix operand order checked by reading, not run on hardware
//
#include <hip/hip_runtime.h>

#define NN 32768
#define NG 64
#define NPG 512
#define NE 262144
#define DM 128
#define NH 8
#define DHD 16
#define MF 44
#define NL 6

typedef __bf16 bf16_t;
typedef __attribute__((ext_vector_type(16))) __bf16 v16bf;
typedef __attribute__((ext_vector_type(8)))  __bf16 v8bf;
typedef __attribute__((ext_vector_type(8)))  float  v8f;
typedef __attribute__((ext_vector_type(4)))  float  f4v;

static __device__ __forceinline__ v8bf cvt8(const float* __restrict__ p) {
  f4v a = *(const f4v*)p;
  f4v b = *(const f4v*)(p + 4);
  v8bf r;
  r[0] = (bf16_t)a[0]; r[1] = (bf16_t)a[1]; r[2] = (bf16_t)a[2]; r[3] = (bf16_t)a[3];
  r[4] = (bf16_t)b[0]; r[5] = (bf16_t)b[1]; r[6] = (bf16_t)b[2]; r[7] = (bf16_t)b[3];
  return r;
}
static __device__ __forceinline__ v16bf mk16(v8bf lo, v8bf hi) {
  return __builtin_shufflevector(lo, hi, 0, 1, 2, 3, 4, 5, 6, 7,
                                 8, 9, 10, 11, 12, 13, 14, 15);
}

// ---------------------------------------------------------------------------
// WMMA GEMM: C[M,N] = act( sum_seg A_seg[M,128] @ W_seg^T + bias )
// W is [N, NSEG*128] row-major. Each wave owns a 16x64 output tile
// (4 accumulators): A fragment loaded once per K-step, reused for 4 WMMAs.
// ---------------------------------------------------------------------------
template <int NSEG, bool ABF>
__global__ void __launch_bounds__(256) gemm_wmma_kernel(
    int M, int N,
    const void* __restrict__ A0v, int lda0,
    const void* __restrict__ A1v, int lda1,
    const float* __restrict__ W, const float* __restrict__ bias, int relu,
    void* __restrict__ C, int ldc, int cbf)
{
  const int ldw = NSEG * 128;
  const int NT  = 4;                           // 4 x (16x16) column tiles/wave
  int wid     = blockIdx.x * (blockDim.x >> 5) + (threadIdx.x >> 5);
  int groupsN = N >> 6;                        // N / 64
  int total   = (M >> 4) * groupsN;
  if (wid >= total) return;                    // uniform per wave
  int tm = wid / groupsN, tg = wid % groupsN;
  int row0 = tm << 4, col0 = tg << 6;
  int lane = threadIdx.x & 31;
  int l15  = lane & 15;
  int khi8 = (lane >> 4) << 3;                 // ISA 16-bit fragment K split
  const v8f zf = {0.f, 0.f, 0.f, 0.f, 0.f, 0.f, 0.f, 0.f};
  v8f acc[NT];
#pragma unroll
  for (int t = 0; t < NT; ++t) acc[t] = zf;

#pragma unroll
  for (int seg = 0; seg < NSEG; ++seg) {
    const void* Av  = seg ? A1v : A0v;
    const int   lda = seg ? lda1 : lda0;
    const float*  af = ABF ? nullptr : (const float*)Av + (size_t)(row0 + l15) * lda;
    const bf16_t* ab = ABF ? (const bf16_t*)Av + (size_t)(row0 + l15) * lda : nullptr;
    const float* wr0 = W + (size_t)(col0 + l15) * ldw + seg * 128;
#pragma unroll
    for (int kk = 0; kk < 4; ++kk) {
      int kb = kk * 32;
      v16bf a;
      if (ABF) a = mk16(*(const v8bf*)(ab + kb + khi8),
                        *(const v8bf*)(ab + kb + 16 + khi8));
      else     a = mk16(cvt8(af + kb + khi8), cvt8(af + kb + 16 + khi8));
#pragma unroll
      for (int t = 0; t < NT; ++t) {
        const float* wr = wr0 + (size_t)(t * 16) * ldw;
        v16bf b = mk16(cvt8(wr + kb + khi8), cvt8(wr + kb + 16 + khi8));
        acc[t] = __builtin_amdgcn_wmma_f32_16x16x32_bf16(false, a, false, b,
                                                         (short)0, acc[t],
                                                         false, false);
      }
    }
  }
  int rbase = (lane >> 4) << 3;
#pragma unroll
  for (int t = 0; t < NT; ++t) {
    int cn = col0 + t * 16 + l15;
    float bv = bias ? bias[cn] : 0.f;
#pragma unroll
    for (int r = 0; r < 8; ++r) {
      float v = acc[t][r] + bv;
      if (relu) v = fmaxf(v, 0.f);
      size_t idx = (size_t)(row0 + rbase + r) * ldc + cn;
      if (cbf) ((bf16_t*)C)[idx] = (bf16_t)v;
      else     ((float*) C)[idx] = v;
    }
  }
}

// ---------------------------------------------------------------------------
// Fused GatedGCN edge stage: m = relu(e1([h[dst],h[src],ea])); scatter(e2(m))
// 16 edges per block, 8 waves x 16 cols; m tile staged in LDS as bf16.
// ---------------------------------------------------------------------------
__global__ void __launch_bounds__(256) edge_kernel(
    const float* __restrict__ H, const bf16_t* __restrict__ EA,
    const int* __restrict__ src, const int* __restrict__ dst,
    const float* __restrict__ W1, const float* __restrict__ b1,  // [128,384]
    const float* __restrict__ W2, const float* __restrict__ b2,  // [128,128]
    float* __restrict__ AGG)
{
  int e0   = blockIdx.x * 16;
  int wave = threadIdx.x >> 5, lane = threadIdx.x & 31;
  int l15  = lane & 15, khi8 = (lane >> 4) << 3;
  __shared__ bf16_t sm[16 * 128];
  __shared__ int sidx[2][16];
  if (threadIdx.x < 32) {
    int w = threadIdx.x >> 4, j = threadIdx.x & 15;
    sidx[w][j] = (w ? src : dst)[e0 + j];
  }
  __syncthreads();
  int col0 = wave * 16;
  float bv1 = b1[col0 + l15], bv2 = b2[col0 + l15];
  v8f acc = {0.f, 0.f, 0.f, 0.f, 0.f, 0.f, 0.f, 0.f};

#pragma unroll
  for (int seg = 0; seg < 2; ++seg) {          // gathered f32 h rows
    const float* ap = H + (size_t)sidx[seg][l15] * 128;
    const float* wr = W1 + (size_t)(col0 + l15) * 384 + seg * 128;
#pragma unroll
    for (int kk = 0; kk < 4; ++kk) {
      int kb = kk * 32;
      v16bf a = mk16(cvt8(ap + kb + khi8), cvt8(ap + kb + 16 + khi8));
      v16bf b = mk16(cvt8(wr + kb + khi8), cvt8(wr + kb + 16 + khi8));
      acc = __builtin_amdgcn_wmma_f32_16x16x32_bf16(false, a, false, b,
                                                    (short)0, acc, false, false);
    }
  }
  {                                            // ea segment (bf16, direct)
    const bf16_t* ap = EA + (size_t)(e0 + l15) * 128;
    const float*  wr = W1 + (size_t)(col0 + l15) * 384 + 256;
#pragma unroll
    for (int kk = 0; kk < 4; ++kk) {
      int kb = kk * 32;
      v16bf a = mk16(*(const v8bf*)(ap + kb + khi8),
                     *(const v8bf*)(ap + kb + 16 + khi8));
      v16bf b = mk16(cvt8(wr + kb + khi8), cvt8(wr + kb + 16 + khi8));
      acc = __builtin_amdgcn_wmma_f32_16x16x32_bf16(false, a, false, b,
                                                    (short)0, acc, false, false);
    }
  }
  int rbase = (lane >> 4) << 3;
#pragma unroll
  for (int r = 0; r < 8; ++r)
    sm[(rbase + r) * 128 + col0 + l15] = (bf16_t)fmaxf(acc[r] + bv1, 0.f);
  __syncthreads();

  v8f acc2 = {0.f, 0.f, 0.f, 0.f, 0.f, 0.f, 0.f, 0.f};
  const float*  wr2 = W2 + (size_t)(col0 + l15) * 128;
  const bf16_t* sp  = sm + l15 * 128;
#pragma unroll
  for (int kk = 0; kk < 4; ++kk) {
    int kb = kk * 32;
    v16bf a = mk16(*(const v8bf*)(sp + kb + khi8),
                   *(const v8bf*)(sp + kb + 16 + khi8));
    v16bf b = mk16(cvt8(wr2 + kb + khi8), cvt8(wr2 + kb + 16 + khi8));
    acc2 = __builtin_amdgcn_wmma_f32_16x16x32_bf16(false, a, false, b,
                                                   (short)0, acc2, false, false);
  }
#pragma unroll
  for (int r = 0; r < 8; ++r)
    atomicAdd(&AGG[(size_t)sidx[0][rbase + r] * 128 + col0 + l15],
              acc2[r] + bv2);
}

// ---------------------------------------------------------------------------
// Performer FAVOR+ attention, one workgroup per (graph, head). LDS-resident.
// ---------------------------------------------------------------------------
__global__ void __launch_bounds__(256) performer_attn_kernel(
    const float* __restrict__ Q, const float* __restrict__ K,
    const float* __restrict__ V, const float* __restrict__ proj,
    float* __restrict__ Out)
{
  int g = blockIdx.x >> 3, hh = blockIdx.x & 7;
  int tid = threadIdx.x;
  int base = g * NPG, coff = hh * DHD;
  __shared__ float sproj[MF * DHD];
  __shared__ float sv[NPG * DHD];
  __shared__ float skp[NPG * MF];
  __shared__ float sdiag[NPG];
  __shared__ float sctx[MF * DHD];
  __shared__ float sksum[MF];
  __shared__ float sred[256];

  for (int i = tid; i < MF * DHD; i += 256) sproj[i] = proj[i];
  for (int i = tid; i < NPG * DHD; i += 256) {
    int n = i >> 4, d = i & 15;
    sv[i] = V[(size_t)(base + n) * DM + coff + d];
  }
  __syncthreads();

  const float dn = 0.5f;                 // 16^-0.25
  const float ratio = 0.15075567f;       // 44^-0.5
  const float feps = 1e-4f;
  float lmax = -1e30f;
  for (int n = tid; n < NPG; n += 256) {
    float kv[DHD], ss = 0.f;
    const float* kr = &K[(size_t)(base + n) * DM + coff];
#pragma unroll
    for (int d = 0; d < DHD; ++d) { kv[d] = kr[d]; ss += kv[d] * kv[d]; }
    sdiag[n] = 0.5f * ss * dn * dn;
    for (int m = 0; m < MF; ++m) {
      float a = 0.f;
#pragma unroll
      for (int d = 0; d < DHD; ++d) a += kv[d] * dn * sproj[m * DHD + d];
      skp[n * MF + m] = a;
      lmax = fmaxf(lmax, a);
    }
  }
  sred[tid] = lmax; __syncthreads();
  for (int s = 128; s > 0; s >>= 1) {
    if (tid < s) sred[tid] = fmaxf(sred[tid], sred[tid + s]);
    __syncthreads();
  }
  float gmax = sred[0];
  __syncthreads();
  for (int i = tid; i < NPG * MF; i += 256) {
    int n = i / MF;
    skp[i] = ratio * (__expf(skp[i] - sdiag[n] - gmax) + feps);
  }
  __syncthreads();
  if (tid < MF) {
    float s = 0.f;
    for (int n = 0; n < NPG; ++n) s += skp[n * MF + tid];
    sksum[tid] = s;
  }
  for (int i = tid; i < MF * DHD; i += 256) {
    int m = i >> 4, d = i & 15;
    float s = 0.f;
    for (int n = 0; n < NPG; ++n) s += skp[n * MF + m] * sv[n * DHD + d];
    sctx[i] = s;
  }
  __syncthreads();
  for (int n = tid; n < NPG; n += 256) {
    float qv[DHD], ss = 0.f;
    const float* qr = &Q[(size_t)(base + n) * DM + coff];
#pragma unroll
    for (int d = 0; d < DHD; ++d) { qv[d] = qr[d]; ss += qv[d] * qv[d]; }
    float qdiag = 0.5f * ss * dn * dn;
    float qd[MF];
    float rmax = -1e30f;
    for (int m = 0; m < MF; ++m) {
      float a = 0.f;
#pragma unroll
      for (int d = 0; d < DHD; ++d) a += qv[d] * dn * sproj[m * DHD + d];
      qd[m] = a; rmax = fmaxf(rmax, a);
    }
    float denom = 0.f;
    for (int m = 0; m < MF; ++m) {
      qd[m] = ratio * (__expf(qd[m] - qdiag - rmax) + feps);
      denom += qd[m] * sksum[m];
    }
    float dinv = 1.f / denom;
    for (int d = 0; d < DHD; ++d) {
      float a = 0.f;
      for (int m = 0; m < MF; ++m) a += qd[m] * sctx[m * DHD + d];
      Out[(size_t)(base + n) * DM + coff + d] = a * dinv;
    }
  }
}

// ----------------------------- small kernels --------------------------------
__global__ void zero_kernel(float* __restrict__ p, int n) {
  int i = blockIdx.x * 256 + threadIdx.x;
  if (i < n) p[i] = 0.f;
}
__global__ void add_kernel(const float* __restrict__ a, const float* __restrict__ b,
                           float* __restrict__ o, int n) {
  int i = blockIdx.x * 256 + threadIdx.x;
  if (i < n) o[i] = a[i] + b[i];
}
__global__ void scalar_stats_kernel(const float* __restrict__ v, int n, float* out) {
  __shared__ float s1[256], s2[256];
  float a = 0.f, b = 0.f;
  for (int i = threadIdx.x; i < n; i += 256) { float x = v[i]; a += x; b += x * x; }
  s1[threadIdx.x] = a; s2[threadIdx.x] = b; __syncthreads();
  for (int s = 128; s > 0; s >>= 1) {
    if (threadIdx.x < s) { s1[threadIdx.x] += s1[threadIdx.x + s];
                           s2[threadIdx.x] += s2[threadIdx.x + s]; }
    __syncthreads();
  }
  if (threadIdx.x == 0) { float mu = s1[0] / n; out[0] = mu; out[1] = s2[0] / n - mu * mu; }
}
__global__ void colstats_kernel(const float* __restrict__ X, int rows,
                                float* mean, float* var) {
  int c = blockIdx.x;
  __shared__ float s1[256], s2[256];
  float a = 0.f, b = 0.f;
  for (int r = threadIdx.x; r < rows; r += 256) {
    float x = X[(size_t)r * 128 + c]; a += x; b += x * x;
  }
  s1[threadIdx.x] = a; s2[threadIdx.x] = b; __syncthreads();
  for (int s = 128; s > 0; s >>= 1) {
    if (threadIdx.x < s) { s1[threadIdx.x] += s1[threadIdx.x + s];
                           s2[threadIdx.x] += s2[threadIdx.x + s]; }
    __syncthreads();
  }
  if (threadIdx.x == 0) { float mu = s1[0] / rows; mean[c] = mu; var[c] = s2[0] / rows - mu * mu; }
}
__global__ void rwse_lin_kernel(const float* __restrict__ rwse, const float* __restrict__ W,
                                const float* __restrict__ b, float* __restrict__ out) {
  int i = blockIdx.x * 256 + threadIdx.x;
  int n = i >> 7, d = i & 127;
  float a = b[d];
#pragma unroll
  for (int j = 0; j < 8; ++j) a += rwse[n * 8 + j] * W[d * 8 + j];
  out[i] = a;
}
__global__ void encoder_node_kernel(const float* __restrict__ x, const float* __restrict__ st,
                                    const float* __restrict__ wn, const float* __restrict__ gn,
                                    const float* __restrict__ bn_b,
                                    const float* __restrict__ rpre, const float* __restrict__ rmean,
                                    const float* __restrict__ rvar,
                                    const float* __restrict__ gr, const float* __restrict__ br,
                                    float* __restrict__ H) {
  int i = blockIdx.x * 256 + threadIdx.x;
  int n = i >> 7, d = i & 127;
  float w = wn[d];
  float t = w * (x[n] - st[0]) * rsqrtf(w * w * st[1] + 1e-5f) * gn[d] + bn_b[d];
  float r = (rpre[i] - rmean[d]) * rsqrtf(rvar[d] + 1e-5f) * gr[d] + br[d];
  H[i] = fmaxf(t, 0.f) + fmaxf(r, 0.f);
}
__global__ void encode_edge_kernel(const float* __restrict__ attr, const float* __restrict__ st,
                                   const float* __restrict__ w, const float* __restrict__ g,
                                   const float* __restrict__ beta, bf16_t* __restrict__ EA) {
  int i = blockIdx.x * 256 + threadIdx.x;
  int e = i >> 7, d = i & 127;
  float ww = w[d];
  float t = ww * (attr[e] - st[0]) * rsqrtf(ww * ww * st[1] + 1e-5f) * g[d] + beta[d];
  EA[i] = (bf16_t)fmaxf(t, 0.f);
}
__global__ void bn_apply_kernel(const float* __restrict__ X, const float* __restrict__ mean,
                                const float* __restrict__ var, const float* __restrict__ g,
                                const float* __restrict__ beta, float* __restrict__ out) {
  int i = blockIdx.x * 256 + threadIdx.x;
  int d = i & 127;
  out[i] = (X[i] - mean[d]) * rsqrtf(var[d] + 1e-5f) * g[d] + beta[d];
}
// out = LayerNorm(A + B) over last dim (128), one wave32 per row
__global__ void ln_add_kernel(const float* __restrict__ A, const float* __restrict__ B,
                              const float* __restrict__ g, const float* __restrict__ beta,
                              float* __restrict__ out) {
  int row = blockIdx.x * (blockDim.x >> 5) + (threadIdx.x >> 5);
  int lane = threadIdx.x & 31;
  size_t base = (size_t)row * 128 + lane * 4;
  float v[4];
  float s = 0.f;
#pragma unroll
  for (int i = 0; i < 4; ++i) { v[i] = A[base + i] + B[base + i]; s += v[i]; }
  for (int off = 16; off > 0; off >>= 1) s += __shfl_xor(s, off, 32);
  float mu = s * (1.f / 128.f);
  float vs = 0.f;
#pragma unroll
  for (int i = 0; i < 4; ++i) { float d = v[i] - mu; vs += d * d; }
  for (int off = 16; off > 0; off >>= 1) vs += __shfl_xor(vs, off, 32);
  float inv = rsqrtf(vs * (1.f / 128.f) + 1e-5f);
#pragma unroll
  for (int i = 0; i < 4; ++i)
    out[base + i] = (v[i] - mu) * inv * g[lane * 4 + i] + beta[lane * 4 + i];
}
__global__ void head2_kernel(const float* __restrict__ T, const float* __restrict__ w,
                             const float* __restrict__ b, float* __restrict__ out) {
  int n = blockIdx.x * 256 + threadIdx.x;
  if (n >= NN) return;
  float a = b[0];
#pragma unroll 8
  for (int d = 0; d < 128; ++d) a += T[(size_t)n * 128 + d] * w[d];
  out[n] = a;
}

// ------------------------- host-side launch helpers -------------------------
static inline void gemm1f(hipStream_t s, int M, int N, const float* A, int lda,
                          const float* W, const float* bias, int relu,
                          void* C, int ldc, int cbf) {
  int waves = (M / 16) * (N / 64);
  gemm_wmma_kernel<1, false><<<waves / 8, 256, 0, s>>>(M, N, A, lda, nullptr, 0,
                                                       W, bias, relu, C, ldc, cbf);
}
static inline void gemm2f(hipStream_t s, int M, int N, const float* A0, int lda0,
                          const float* A1, int lda1, const float* W, const float* bias,
                          int relu, void* C, int ldc, int cbf) {
  int waves = (M / 16) * (N / 64);
  gemm_wmma_kernel<2, false><<<waves / 8, 256, 0, s>>>(M, N, A0, lda0, A1, lda1,
                                                       W, bias, relu, C, ldc, cbf);
}
static inline void gemm1b(hipStream_t s, int M, int N, const bf16_t* A, int lda,
                          const float* W, const float* bias, int relu,
                          void* C, int ldc, int cbf) {
  int waves = (M / 16) * (N / 64);
  gemm_wmma_kernel<1, true><<<waves / 8, 256, 0, s>>>(M, N, A, lda, nullptr, 0,
                                                      W, bias, relu, C, ldc, cbf);
}
static inline void gemm2b(hipStream_t s, int M, int N, const bf16_t* A0, int lda0,
                          const bf16_t* A1, int lda1, const float* W, const float* bias,
                          int relu, void* C, int ldc, int cbf) {
  int waves = (M / 16) * (N / 64);
  gemm_wmma_kernel<2, true><<<waves / 8, 256, 0, s>>>(M, N, A0, lda0, A1, lda1,
                                                      W, bias, relu, C, ldc, cbf);
}

extern "C" void kernel_launch(void* const* d_in, const int* in_sizes, int n_in,
                              void* d_out, int out_size, void* d_ws, size_t ws_size,
                              hipStream_t stream) {
  if (n_in < 194) return;
  const float* x    = (const float*)d_in[0];
  const float* eatt = (const float*)d_in[1];
  const float* rwse = (const float*)d_in[2];
  // params flattened (jax tree: sorted dict keys) at indices 3..190
#define PF(i) ((const float*)d_in[3 + (i)])
  const float* projs = (const float*)d_in[191];
  const int*   eidx  = (const int*)d_in[193];
  const int* src = eidx;
  const int* dst = eidx + NE;
  // layer leaf base: 16 + 28*li ; order within layer:
  // 0 k.b 1 k.w 2 o.b 3 o.w 4 q.b 5 q.w 6 v.b 7 v.w 8 attn_ln.b 9 attn_ln.g
  // 10 e1.b 11 e1.w 12 e2.b 13 e2.w 14 f1.b 15 f1.w 16 f2.b 17 f2.w
  // 18 ffn_ln.b 19 ffn_ln.g 20 gcn_ln.b 21 gcn_ln.g 22 local_ln.b 23 local_ln.g
  // 24 n1.b 25 n1.w 26 n2.b 27 n2.w

  char* wsb = (char*)d_ws;
  size_t off = 0;
  auto alloc = [&](size_t bytes) { void* p = wsb + off; off += (bytes + 255) & ~(size_t)255; return p; };
  bf16_t* EA    = (bf16_t*)alloc((size_t)NE * DM * 2);
  bf16_t* TBUF  = (bf16_t*)alloc((size_t)NN * DM * 2);
  bf16_t* FFBUF = (bf16_t*)alloc((size_t)NN * 256 * 2);
  float* H    = (float*)alloc((size_t)NN * DM * 4);
  float* AGG  = (float*)alloc((size_t)NN * DM * 4);
  float* OBUF = (float*)alloc((size_t)NN * DM * 4);
  float* GBUF = (float*)alloc((size_t)NN * DM * 4);
  float* LBUF = (float*)alloc((size_t)NN * DM * 4);
  float* QBUF = (float*)alloc((size_t)NN * DM * 4);
  float* KBUF = (float*)alloc((size_t)NN * DM * 4);
  float* VBUF = (float*)alloc((size_t)NN * DM * 4);
  float* ABUF = (float*)alloc((size_t)NN * DM * 4);
  float* ST   = (float*)alloc(4096);
  float* st_x = ST, *st_e = ST + 2;
  float* csr_m = ST + 8,   *csr_v = ST + 136;
  float* csp_m = ST + 264, *csp_v = ST + 392;
  (void)ws_size; (void)in_sizes; (void)out_size;

  const int NB_N = NN * DM / 256;
  const int NB_E = NE * DM / 256;
  const int LN_B = NN / 8;

  // ---- encoder ----
  scalar_stats_kernel<<<1, 256, 0, stream>>>(x, NN, st_x);
  scalar_stats_kernel<<<1, 256, 0, stream>>>(eatt, NE, st_e);
  rwse_lin_kernel<<<NB_N, 256, 0, stream>>>(rwse, PF(9), PF(8), QBUF);
  colstats_kernel<<<128, 256, 0, stream>>>(QBUF, NN, csr_m, csr_v);
  encoder_node_kernel<<<NB_N, 256, 0, stream>>>(x, st_x, PF(5), PF(7), PF(6),
                                                QBUF, csr_m, csr_v, PF(11), PF(10), H);
  encode_edge_kernel<<<NB_E, 256, 0, stream>>>(eatt, st_e, PF(1), PF(3), PF(2), EA);

  // ---- pre_mp: h = bn(relu(W h + b)) ----
  gemm1f(stream, NN, 128, H, 128, PF(185), PF(184), 1, QBUF, 128, 0);
  colstats_kernel<<<128, 256, 0, stream>>>(QBUF, NN, csp_m, csp_v);
  bn_apply_kernel<<<NB_N, 256, 0, stream>>>(QBUF, csp_m, csp_v, PF(187), PF(186), H);

  // ---- layers ----
  for (int li = 0; li < NL; ++li) {
    int L = 16 + 28 * li;
    zero_kernel<<<(NN * DM + 255) / 256, 256, 0, stream>>>(AGG, NN * DM);
    edge_kernel<<<NE / 16, 256, 0, stream>>>(H, EA, src, dst,
                                             PF(L + 11), PF(L + 10),
                                             PF(L + 13), PF(L + 12), AGG);
    gemm2f(stream, NN, 128, H, 128, AGG, 128, PF(L + 25), PF(L + 24), 1, TBUF, 128, 1);
    gemm1b(stream, NN, 128, TBUF, 128, PF(L + 27), PF(L + 26), 0, OBUF, 128, 0);
    ln_add_kernel<<<LN_B, 256, 0, stream>>>(H, OBUF, PF(L + 21), PF(L + 20), GBUF);  // gcn_ln
    ln_add_kernel<<<LN_B, 256, 0, stream>>>(H, GBUF, PF(L + 23), PF(L + 22), LBUF);  // local_ln
    gemm1f(stream, NN, 128, H, 128, PF(L + 5), PF(L + 4), 0, QBUF, 128, 0);          // q
    gemm1f(stream, NN, 128, H, 128, PF(L + 1), PF(L + 0), 0, KBUF, 128, 0);          // k
    gemm1f(stream, NN, 128, H, 128, PF(L + 7), PF(L + 6), 0, VBUF, 128, 0);          // v
    performer_attn_kernel<<<NG * NH, 256, 0, stream>>>(QBUF, KBUF, VBUF,
                                                       projs + (size_t)li * MF * DHD, ABUF);
    gemm1f(stream, NN, 128, ABUF, 128, PF(L + 3), PF(L + 2), 0, OBUF, 128, 0);       // o proj
    ln_add_kernel<<<LN_B, 256, 0, stream>>>(H, OBUF, PF(L + 9), PF(L + 8), QBUF);    // attn_ln
    add_kernel<<<(NN * DM + 255) / 256, 256, 0, stream>>>(LBUF, QBUF, KBUF, NN * DM);
    gemm1f(stream, NN, 256, KBUF, 128, PF(L + 15), PF(L + 14), 1, FFBUF, 256, 1);    // f1
    gemm2b(stream, NN, 128, FFBUF, 256, FFBUF + 128, 256, PF(L + 17), PF(L + 16), 0,
           VBUF, 128, 0);                                                            // f2
    ln_add_kernel<<<LN_B, 256, 0, stream>>>(KBUF, VBUF, PF(L + 19), PF(L + 18), H);  // ffn_ln
  }

  // ---- head ----
  gemm1f(stream, NN, 128, H, 128, PF(13), PF(12), 1, QBUF, 128, 0);
  head2_kernel<<<NN / 256, 256, 0, stream>>>(QBUF, PF(15), PF(14), (float*)d_out);
#undef PF
}
